// ConvLayer_13967233646751
// MI455X (gfx1250) — compile-verified, matrix operand order checked by reading, not run on hardware
//
#include <hip/hip_runtime.h>

typedef __attribute__((ext_vector_type(16))) __bf16 v16bf;
typedef __attribute__((ext_vector_type(8)))  float  v8f;

#define NB 16
#define NC 76
#define NP 1024
#define NS 32
#define NPTS (NB * NP)
#define NPAIRS (NPTS / 2)          // two consecutive points per block-iteration
#define COUT 131
#define CSTRIDE ((size_t)(NP * NS))   // elements between input channels

// float -> bf16 bits, round-to-nearest-even
__device__ __forceinline__ unsigned short f2bf(float f) {
  unsigned int u = __float_as_uint(f);
  u += 0x7FFFu + ((u >> 16) & 1u);
  return (unsigned short)(u >> 16);
}
__device__ __forceinline__ __bf16 bfbits(unsigned short b) {
  return __builtin_bit_cast(__bf16, b);
}
__device__ __forceinline__ unsigned int pack2(unsigned short lo, unsigned short hi) {
  return (unsigned int)lo | ((unsigned int)hi << 16);
}
// K index inside a 16x32 bf16 A/B fragment for VGPR r, half-wave h (ISA 7.12.2)
__device__ __forceinline__ int frag_k(int r, int h) {
  return (r < 4 ? 2 * r : 16 + 2 * (r - 4)) + 8 * h;
}

__global__ __launch_bounds__(256) void fused_pointmlp(
    const float* __restrict__ input, const float* __restrict__ normal,
    const float* __restrict__ w1, const float* __restrict__ b1,
    const float* __restrict__ w2, const float* __restrict__ b2,
    float* __restrict__ out) {
  // bf16 pairs packed in dwords: x tile [(q*32+s)][cpair], K padded 73->96
  __shared__ unsigned int xu[64 * 48];   // 12 KB
  // h1 tile [(n)][rowpair], 128 rows
  __shared__ unsigned int hu[64 * 64];   // 16 KB
  __shared__ float geo[24];              // per point q: R row-major (9) + azi_u (3)
  __shared__ float gtmp[6];

  const int tid  = threadIdx.x;
  const int wave = tid >> 5;
  const int lane = tid & 31;
  const int half = lane >> 4;
  const int l16  = lane & 15;

  // ---- Build weight A-fragments once (bf16 16x32 tiles, rows = out channels) ----
  v16bf A1[3];   // W1: 128x73 padded to K=96
  v16bf A2[4];   // W2: 128x128
  {
    const int m = wave * 16 + l16;          // output row this lane owns
    #pragma unroll
    for (int ks = 0; ks < 3; ++ks) {
      #pragma unroll
      for (int r = 0; r < 8; ++r) {
        const int k  = ks * 32 + frag_k(r, half);
        const float f0 = (k     < 73) ? w1[m * 73 + k]     : 0.f;
        const float f1 = (k + 1 < 73) ? w1[m * 73 + k + 1] : 0.f;
        A1[ks][2 * r]     = bfbits(f2bf(f0));
        A1[ks][2 * r + 1] = bfbits(f2bf(f1));
      }
    }
    #pragma unroll
    for (int ks = 0; ks < 4; ++ks) {
      #pragma unroll
      for (int r = 0; r < 8; ++r) {
        const int k = ks * 32 + frag_k(r, half);
        A2[ks][2 * r]     = bfbits(f2bf(w2[m * 128 + k]));
        A2[ks][2 * r + 1] = bfbits(f2bf(w2[m * 128 + k + 1]));
      }
    }
  }
  float bias1[8], bias2[8];
  #pragma unroll
  for (int g = 0; g < 8; ++g) {
    const int row = wave * 16 + g + 8 * half;   // D-matrix row for acc[g]
    bias1[g] = b1[row];
    bias2[g] = b2[row];
  }

  for (int pair = blockIdx.x; pair < NPAIRS; pair += gridDim.x) {
    const int pid0 = pair * 2;                        // even -> same batch b
    const int b0 = pid0 >> 10;
    const int p0 = pid0 & (NP - 1);
    const size_t base0 = ((size_t)b0 * NC * NP + p0) * NS;  // point q: +q*NS; chan c: +c*CSTRIDE

    // ---- prefetch next pair's input channels into cache (global_prefetch) ----
    {
      const int npair = pair + gridDim.x;
      if (npair < NPAIRS && tid < 2 * NC) {
        const int pn = npair * 2;
        const size_t nb = ((size_t)(pn >> 10) * NC * NP + (pn & (NP - 1))) * NS;
        __builtin_prefetch(input + nb + (size_t)(tid >> 1) * CSTRIDE + (tid & 1) * 32, 0, 0);
      }
    }

    // ---- azi = mean_{s=1..31} rel ; one wave per (point, component) ----
    if (wave < 6) {
      const int q = wave / 3;
      const int j = wave - 3 * q;
      float v = (lane >= 1)
                  ? input[base0 + (size_t)q * NS + (size_t)(6 + j) * CSTRIDE + lane]
                  : 0.f;
      #pragma unroll
      for (int off = 16; off > 0; off >>= 1) v += __shfl_xor(v, off, 32);
      if (lane == 0) gtmp[q * 3 + j] = v * (1.0f / 31.0f);
    }
    __syncthreads();

    if (tid < 2) {
      const int q = tid;
      const int pid = pid0 + q;
      float a0 = gtmp[q * 3 + 0], a1 = gtmp[q * 3 + 1], a2 = gtmp[q * 3 + 2];
      float n0 = normal[(size_t)pid * 3 + 0];
      float n1 = normal[(size_t)pid * 3 + 1];
      float n2 = normal[(size_t)pid * 3 + 2];
      float ln = sqrtf(n0 * n0 + n1 * n1 + n2 * n2) + 1e-8f;
      n0 /= ln; n1 /= ln; n2 /= ln;
      float la = sqrtf(a0 * a0 + a1 * a1 + a2 * a2) + 1e-8f;
      a0 /= la; a1 /= la; a2 /= la;
      const float d = a0 * n0 + a1 * n1 + a2 * n2;
      float x0 = a0 - d * n0, x1 = a1 - d * n1, x2 = a2 - d * n2;
      float lx = sqrtf(x0 * x0 + x1 * x1 + x2 * x2) + 1e-8f;
      x0 /= lx; x1 /= lx; x2 /= lx;
      const float y0 = n1 * x2 - n2 * x1;
      const float y1 = n2 * x0 - n0 * x2;
      const float y2 = n0 * x1 - n1 * x0;
      float* g = geo + q * 12;
      g[0] = x0; g[1] = x1; g[2] = x2;
      g[3] = y0; g[4] = y1; g[5] = y2;
      g[6] = n0; g[7] = n1; g[8] = n2;
      g[9] = a0; g[10] = a1; g[11] = a2;
      const int bq = pid >> 10, pq = pid & (NP - 1);
      out[((size_t)bq * COUT + 0) * NP + pq] = a0;
      out[((size_t)bq * COUT + 1) * NP + pq] = a1;
      out[((size_t)bq * COUT + 2) * NP + pq] = a2;
    }
    __syncthreads();

    // ---- build x (96 x 64) into LDS as packed bf16 pairs ----
    #pragma unroll
    for (int q = 0; q < 2; ++q) {
      float R[9], au[3];
      #pragma unroll
      for (int i = 0; i < 9; ++i) R[i] = geo[q * 12 + i];
      #pragma unroll
      for (int i = 0; i < 3; ++i) au[i] = geo[q * 12 + 9 + i];
      const size_t bq = base0 + (size_t)q * NS;

      for (int idx = tid; idx < 48 * NS; idx += 256) {
        const int s  = idx & 31;
        const int cp = idx >> 5;          // channel pair 0..47
        float v01[2];
        #pragma unroll
        for (int e = 0; e < 2; ++e) {
          const int c = 2 * cp + e;
          float v = 0.f;
          if (c < 3) {                       // align(rel), rel = ch 6..8
            const float r0 = input[bq + 6 * CSTRIDE + s];
            const float r1 = input[bq + 7 * CSTRIDE + s];
            const float r2 = input[bq + 8 * CSTRIDE + s];
            v = R[c * 3 + 0] * r0 + R[c * 3 + 1] * r1 + R[c * 3 + 2] * r2;
          } else if (c < 67) {               // feats = ch 12..75
            v = input[bq + (size_t)(12 + c - 3) * CSTRIDE + s];
          } else if (c < 70) {               // align(other_normal), ch 3..5
            const int j = c - 67;
            const float o0 = input[bq + 3 * CSTRIDE + s];
            const float o1 = input[bq + 4 * CSTRIDE + s];
            const float o2 = input[bq + 5 * CSTRIDE + s];
            v = R[j * 3 + 0] * o0 + R[j * 3 + 1] * o1 + R[j * 3 + 2] * o2;
          } else if (c < 73) {               // dir_dif = align(azi_u - other_dir)
            const int j = c - 70;
            const float d0 = au[0] - input[bq + 9 * CSTRIDE + s];
            const float d1 = au[1] - input[bq + 10 * CSTRIDE + s];
            const float d2 = au[2] - input[bq + 11 * CSTRIDE + s];
            v = R[j * 3 + 0] * d0 + R[j * 3 + 1] * d1 + R[j * 3 + 2] * d2;
          }                                  // c in [73,96): zero pad
          v01[e] = v;
        }
        xu[(q * 32 + s) * 48 + cp] = pack2(f2bf(v01[0]), f2bf(v01[1]));
      }
    }
    __syncthreads();

    // ---- layer 1: h1 = relu(W1 x + b1); 4 N-tiles x 3 K-steps ----
    #pragma unroll
    for (int nt = 0; nt < 4; ++nt) {
      const int n = nt * 16 + l16;          // global column 0..63
      v16bf Bf[3];
      #pragma unroll
      for (int ks = 0; ks < 3; ++ks) {
        #pragma unroll
        for (int r = 0; r < 8; ++r) {
          const unsigned int pr = xu[n * 48 + ((ks * 32 + frag_k(r, half)) >> 1)];
          Bf[ks][2 * r]     = bfbits((unsigned short)(pr & 0xFFFFu));
          Bf[ks][2 * r + 1] = bfbits((unsigned short)(pr >> 16));
        }
      }
      v8f acc = {};
      #pragma unroll
      for (int ks = 0; ks < 3; ++ks)
        acc = __builtin_amdgcn_wmma_f32_16x16x32_bf16(
            false, A1[ks], false, Bf[ks], (short)0, acc, false, false);
      #pragma unroll
      for (int g = 0; g < 8; g += 2) {
        const unsigned short e0 = f2bf(fmaxf(acc[g]     + bias1[g],     0.f));
        const unsigned short e1 = f2bf(fmaxf(acc[g + 1] + bias1[g + 1], 0.f));
        // row = wave*16 + g + 8*half (even), pair index = row/2
        hu[n * 64 + wave * 8 + 4 * half + (g >> 1)] = pack2(e0, e1);
      }
    }
    __syncthreads();

    // ---- layer 2: 4 N-tiles x 4 K-steps, then S-max-pool per point ----
    v8f accs[4];
    #pragma unroll
    for (int nt = 0; nt < 4; ++nt) {
      const int n = nt * 16 + l16;
      v16bf Bf[4];
      #pragma unroll
      for (int ks = 0; ks < 4; ++ks) {
        #pragma unroll
        for (int r = 0; r < 8; ++r) {
          const unsigned int pr = hu[n * 64 + ((ks * 32 + frag_k(r, half)) >> 1)];
          Bf[ks][2 * r]     = bfbits((unsigned short)(pr & 0xFFFFu));
          Bf[ks][2 * r + 1] = bfbits((unsigned short)(pr >> 16));
        }
      }
      v8f acc = {};
      #pragma unroll
      for (int ks = 0; ks < 4; ++ks)
        acc = __builtin_amdgcn_wmma_f32_16x16x32_bf16(
            false, A2[ks], false, Bf[ks], (short)0, acc, false, false);
      accs[nt] = acc;
    }
    #pragma unroll
    for (int q = 0; q < 2; ++q) {
      const int pid = pid0 + q;
      const int bq = pid >> 10, pq = pid & (NP - 1);
      #pragma unroll
      for (int g = 0; g < 8; ++g) {
        // max over s of relu(a_s + b) == relu(max_s(a_s) + b)
        float v = fmaxf(fmaxf(accs[2 * q][g], accs[2 * q + 1][g]) + bias2[g], 0.f);
        #pragma unroll
        for (int off = 1; off < 16; off <<= 1)
          v = fmaxf(v, __shfl_xor(v, off, 32));   // reduce over the 16 N-lanes
        if (l16 == 0) {
          const int row = wave * 16 + g + 8 * half;
          out[((size_t)bq * COUT + 3 + row) * NP + pq] = v;
        }
      }
    }
  }
}

extern "C" void kernel_launch(void* const* d_in, const int* in_sizes, int n_in,
                              void* d_out, int out_size, void* d_ws, size_t ws_size,
                              hipStream_t stream) {
  (void)in_sizes; (void)n_in; (void)d_ws; (void)ws_size; (void)out_size;
  const float* input  = (const float*)d_in[0];
  const float* normal = (const float*)d_in[1];
  const float* w1     = (const float*)d_in[2];
  const float* b1     = (const float*)d_in[3];
  const float* w2     = (const float*)d_in[4];
  const float* b2     = (const float*)d_in[5];
  float* out = (float*)d_out;
  // 2048 blocks x 256 threads; each block handles 4 point-pairs (8 points),
  // reusing register-resident bf16 weight fragments across the loop.
  fused_pointmlp<<<2048, 256, 0, stream>>>(input, normal, w1, b1, w2, b2, out);
}
